// Block_70420283785984
// MI455X (gfx1250) — matrix operand order repaired and checked by reading, not compiled, not run on hardware
//
#include <hip/hip_runtime.h>
#include <hip/hip_fp16.h>

typedef _Float16 h16;
typedef __attribute__((ext_vector_type(8)))  _Float16 v8h;
typedef __attribute__((ext_vector_type(16))) _Float16 v16h;
typedef __attribute__((ext_vector_type(8)))  float    v8f;

static constexpr int TT   = 4;     // spiking time steps
static constexpr int BB   = 32;    // batch per step
static constexpr int TB   = 128;   // TT*BB
static constexpr int CC   = 384;
static constexpr int HIDD = 1536;
static constexpr int NN   = 196;   // 14*14
static constexpr int NPAD = 224;   // N padded to multiple of 32 (7 x 32)
static constexpr int NH   = 12;    // heads
static constexpr int HD   = 32;    // head dim

// ---------------- WMMA helpers -------------------------------------------

__device__ __forceinline__ v16h mkfrag(v8h lo, v8h hi_) {
  v16h r;
#pragma unroll
  for (int i = 0; i < 8; ++i) { r[i] = lo[i]; r[i + 8] = hi_[i]; }
  return r;
}

__device__ __forceinline__ v8f wmma16(v16h a, v16h b, v8f c) {
  // D = A(16x32 f16) * B(32x16 f16) + C(16x16 f32)
  return __builtin_amdgcn_wmma_f32_16x16x32_f16(
      /*neg_a=*/false, a, /*neg_b=*/false, b,
      /*c_mod=*/(short)0, c, /*reuse_a=*/false, /*reuse_b=*/false);
}

__device__ __forceinline__ float bnv(float y, const float* __restrict__ p,
                                     int Cdim, int c) {
  // p layout: [gamma, beta, mean, var] each [Cdim]
  float g  = p[c];
  float be = p[Cdim + c];
  float mu = p[2 * Cdim + c];
  float va = p[3 * Cdim + c];
  return (y - mu) * (g * rsqrtf(va + 1e-5f)) + be;
}

// ---------------- conversion / transpose kernels -------------------------

__global__ void kcvt(const float* __restrict__ s, h16* __restrict__ d, int n) {
  int i = blockIdx.x * 256 + threadIdx.x;
  if (i < n) d[i] = (h16)s[i];
}

// xT[b][n][c] = x[b][c][n]  (f32 -> f16, rows n>=NN zero-padded)
__global__ void ktransp(const float* __restrict__ x, h16* __restrict__ xT) {
  int idx = blockIdx.x * 256 + threadIdx.x;
  if (idx >= TB * NPAD * CC) return;
  int c = idx % CC;
  int n = (idx / CC) % NPAD;
  int b = idx / (CC * NPAD);
  float v = (n < NN) ? x[((size_t)b * CC + c) * NN + n] : 0.0f;
  xT[idx] = (h16)v;
}

// ---------------- batched GEMM: Y[b] = A(MxK) * Bt[b]^T ------------------
// A f16 row-major [M][K]; Bt f16 n-major [b][NPAD][K] (zero-padded rows);
// out f16 [b][M][NN]; optional fp32 bias per output row m.
// Wave computes a 32x32 tile: 2 A frags x 2 B frags -> 4 WMMA per K-step.

__global__ __launch_bounds__(128) void kgemm(const h16* __restrict__ A,
                                             const h16* __restrict__ Bt,
                                             const float* __restrict__ bias,
                                             h16* __restrict__ out,
                                             int M, int K) {
  const int b    = blockIdx.z;
  const int n0   = blockIdx.x * 32;
  const int wave = threadIdx.x >> 5;
  const int lane = threadIdx.x & 31;
  const int lm   = lane & 15;
  const int gr   = lane >> 4;
  const int m0   = blockIdx.y * 128 + wave * 32;

  const h16* Ap0 = A + (size_t)(m0 + lm) * K;        // A rows m0+lm
  const h16* Ap1 = Ap0 + (size_t)16 * K;             // A rows m0+16+lm
  const h16* Bp0 = Bt + ((size_t)b * NPAD + n0 + lm) * K;        // col n0+lm
  const h16* Bp1 = Bp0 + (size_t)16 * K;                         // col n0+16+lm

  v8f a00 = {}, a01 = {}, a10 = {}, a11 = {};
  for (int k0 = 0; k0 < K; k0 += 32) {
    v16h fa0 = *(const v16h*)(Ap0 + k0 + gr * 16);
    v16h fa1 = *(const v16h*)(Ap1 + k0 + gr * 16);
    v16h fb0 = *(const v16h*)(Bp0 + k0 + gr * 16);
    v16h fb1 = *(const v16h*)(Bp1 + k0 + gr * 16);
    a00 = wmma16(fa0, fb0, a00);
    a01 = wmma16(fa0, fb1, a01);
    a10 = wmma16(fa1, fb0, a10);
    a11 = wmma16(fa1, fb1, a11);
  }

  const int na = n0 + lm;        // D: lane col = lm (tile ni=0)
  const int nb = n0 + 16 + lm;   // tile ni=1
#pragma unroll
  for (int r = 0; r < 8; ++r) {  // D: row = r + gr*8
    int mA = m0 + r + gr * 8;
    int mB = mA + 16;
    float biA = bias ? bias[mA] : 0.0f;
    float biB = bias ? bias[mB] : 0.0f;
    if (na < NN) {
      out[((size_t)b * M + mA) * NN + na] = (h16)(a00[r] + biA);
      out[((size_t)b * M + mB) * NN + na] = (h16)(a10[r] + biB);
    }
    if (nb < NN) {
      out[((size_t)b * M + mA) * NN + nb] = (h16)(a01[r] + biA);
      out[((size_t)b * M + mB) * NN + nb] = (h16)(a11[r] + biB);
    }
  }
}

// ---------------- fused spiking self-attention ---------------------------
// qs/ks: [b][h][NPAD][HD] f16 spikes (zero-padded rows)
// vT:    [b][h][HD][NPAD] f16 spikes (zero-padded cols)
// obuf:  [b][c=h*32+dd][n] f32 (head-major channel layout of the reference)

__global__ __launch_bounds__(32) void kattn(const h16* __restrict__ qs,
                                            const h16* __restrict__ ks,
                                            const h16* __restrict__ vT,
                                            float* __restrict__ obuf) {
  __shared__ __align__(64) h16 smA[16][NPAD]; // attn tile (f16 exact: k*0.125)
  const int b  = blockIdx.z;
  const int hh = blockIdx.y;
  const int q0 = blockIdx.x * 16;
  const int lane = threadIdx.x & 31;
  const int lm = lane & 15, gr = lane >> 4;

  const size_t bh = (size_t)b * NH + hh;
  // A = q rows (loaded once, reused for 14 WMMAs)
  const h16* qrow = qs + (bh * NPAD + (q0 + lm)) * HD;
  v16h aq = *(const v16h*)(qrow + gr * 16);
  // stage 1: attn(16 x 224) = q(16x32) * k^T(32x224), K = d = 32
#pragma unroll
  for (int kt = 0; kt < 14; ++kt) {
    int krow = kt * 16 + lm;                  // B col -> k row (zero-padded)
    v16h bf = *(const v16h*)(ks + (bh * NPAD + krow) * HD + gr * 16);
    v8f c = {};
    c = wmma16(aq, bf, c);
#pragma unroll
    for (int r = 0; r < 8; ++r)
      smA[r + gr * 8][kt * 16 + lm] = (h16)(c[r] * 0.125f);
  }
  __syncthreads();
  // stage 2: o(16x32) = attn(16x224) * v(224x32), 7 K-steps x 2 col-tiles
  v8f o0 = {}, o1 = {};
#pragma unroll
  for (int k2 = 0; k2 < 7; ++k2) {
    const h16* ar = &smA[lm][k2 * 32];
    v16h af = *(const v16h*)(ar + gr * 16);
    const h16* v0p = vT + (bh * HD + lm)      * NPAD + k2 * 32 + gr * 16;
    const h16* v1p = vT + (bh * HD + 16 + lm) * NPAD + k2 * 32 + gr * 16;
    o0 = wmma16(af, *(const v16h*)v0p, o0);
    o1 = wmma16(af, *(const v16h*)v1p, o1);
  }
#pragma unroll
  for (int r = 0; r < 8; ++r) {
    int n = q0 + r + gr * 8;
    if (n < NN) {
      obuf[((size_t)b * CC + hh * HD + lm)      * NN + n] = o0[r];
      obuf[((size_t)b * CC + hh * HD + 16 + lm) * NN + n] = o1[r];
    }
  }
}

// ---------------- LIF kernels (membrane scan over T=4, b = t*32+j) -------
// All producers of GEMM-B operands write n-major zero-padded layouts.

__global__ __launch_bounds__(256) void klif_qkv(const h16* __restrict__ y,
                                                const float* __restrict__ bnp,
                                                h16* __restrict__ headdst,
                                                h16* __restrict__ vTd,
                                                float* __restrict__ vout,
                                                int isV) {
  int idx = blockIdx.x * 256 + threadIdx.x;
  if (idx >= BB * CC * NPAD) return;
  int n = idx % NPAD;
  int c = (idx / NPAD) % CC;
  int j = idx / (NPAD * CC);
  int hh = c >> 5, dd = c & 31;
  bool valid = (n < NN);
  float mem = 0.0f;
#pragma unroll
  for (int t = 0; t < TT; ++t) {
    int bidx = t * BB + j;
    float s = 0.0f;
    if (valid) {
      float yv = bnv((float)y[((size_t)bidx * CC + c) * NN + n], bnp, CC, c);
      mem += (yv - mem) * 0.5f;
      s = (mem > 1.0f) ? 1.0f : 0.0f;
      mem *= (1.0f - s);
    }
    size_t bh = (size_t)bidx * NH + hh;
    if (isV) {
      vTd[(bh * HD + dd) * NPAD + n] = (h16)s;         // transposed, padded
      if (valid) vout[(bh * NN + n) * HD + dd] = s;    // tuple output v (f32)
    } else {
      headdst[(bh * NPAD + n) * HD + dd] = (h16)s;     // [b][h][n][d] padded
    }
  }
}

// attn LIF (thr 0.5, no BN): obuf f32 [b][c][n] -> sbT f16 [b][n][c] padded
__global__ __launch_bounds__(256) void klif_attn(const float* __restrict__ o,
                                                 h16* __restrict__ sbT) {
  int idx = blockIdx.x * 256 + threadIdx.x;
  if (idx >= BB * CC * NPAD) return;
  int n = idx % NPAD;
  int c = (idx / NPAD) % CC;
  int j = idx / (NPAD * CC);
  bool valid = (n < NN);
  float mem = 0.0f;
#pragma unroll
  for (int t = 0; t < TT; ++t) {
    int bidx = t * BB + j;
    float s = 0.0f;
    if (valid) {
      mem += (o[((size_t)bidx * CC + c) * NN + n] - mem) * 0.5f;
      s = (mem > 0.5f) ? 1.0f : 0.0f;
      mem *= (1.0f - s);
    }
    sbT[((size_t)bidx * NPAD + n) * CC + c] = (h16)s;
  }
}

// proj LIF + attention residual: x1 = x + s  -> out_x (f32) and x1T (f16, padded)
__global__ __launch_bounds__(256) void klif_proj(const h16* __restrict__ y,
                                                 const float* __restrict__ bnp,
                                                 const float* __restrict__ xin,
                                                 float* __restrict__ xout,
                                                 h16* __restrict__ x1T) {
  int idx = blockIdx.x * 256 + threadIdx.x;
  if (idx >= BB * CC * NPAD) return;
  int n = idx % NPAD;
  int c = (idx / NPAD) % CC;
  int j = idx / (NPAD * CC);
  bool valid = (n < NN);
  float mem = 0.0f;
#pragma unroll
  for (int t = 0; t < TT; ++t) {
    int bidx = t * BB + j;
    float x1 = 0.0f;
    if (valid) {
      size_t pos = ((size_t)bidx * CC + c) * NN + n;
      float yv = bnv((float)y[pos], bnp, CC, c);
      mem += (yv - mem) * 0.5f;
      float s = (mem > 1.0f) ? 1.0f : 0.0f;
      mem *= (1.0f - s);
      x1 = xin[pos] + s;
      xout[pos] = x1;
    }
    x1T[((size_t)bidx * NPAD + n) * CC + c] = (h16)x1;
  }
}

// MLP LIF: fc1 -> spikes sT (f16, padded, n-major); fc2 -> accumulate out_x
__global__ __launch_bounds__(256) void klif_mlp(const h16* __restrict__ y,
                                                const float* __restrict__ bnp,
                                                int Cdim,
                                                h16* __restrict__ sT,
                                                float* __restrict__ xadd) {
  int idx = blockIdx.x * 256 + threadIdx.x;
  if (idx >= BB * Cdim * NPAD) return;
  int n = idx % NPAD;
  int c = (idx / NPAD) % Cdim;
  int j = idx / (NPAD * Cdim);
  bool valid = (n < NN);
  float mem = 0.0f;
#pragma unroll
  for (int t = 0; t < TT; ++t) {
    int bidx = t * BB + j;
    float s = 0.0f;
    if (valid) {
      float yv = bnv((float)y[((size_t)bidx * Cdim + c) * NN + n], bnp, Cdim, c);
      mem += (yv - mem) * 0.5f;
      s = (mem > 1.0f) ? 1.0f : 0.0f;
      mem *= (1.0f - s);
    }
    if (sT) sT[((size_t)bidx * NPAD + n) * Cdim + c] = (h16)s;
    else if (valid) xadd[((size_t)bidx * Cdim + c) * NN + n] += s;
  }
}

// ---------------- host orchestration -------------------------------------

extern "C" void kernel_launch(void* const* d_in, const int* in_sizes, int n_in,
                              void* d_out, int out_size, void* d_ws, size_t ws_size,
                              hipStream_t stream) {
  (void)in_sizes; (void)n_in; (void)out_size; (void)ws_size;
  const float* x       = (const float*)d_in[0];
  const float* qkvp_w  = (const float*)d_in[1];
  const float* qkvp_bn = (const float*)d_in[2];
  const float* fc1_w   = (const float*)d_in[3];
  const float* fc1_b   = (const float*)d_in[4];
  const float* fc1_bn  = (const float*)d_in[5];
  const float* fc2_w   = (const float*)d_in[6];
  const float* fc2_b   = (const float*)d_in[7];
  const float* fc2_bn  = (const float*)d_in[8];

  float* out_x = (float*)d_out;
  float* out_v = out_x + (size_t)TB * CC * NN;

  char* wsp = (char*)d_ws;
  size_t off = 0;
  auto alloc = [&](size_t bytes) -> void* {
    void* p = wsp + off;
    off = (off + bytes + 255) & ~(size_t)255;
    return p;
  };

  h16*   xT   = (h16*)alloc(sizeof(h16) * (size_t)TB * NPAD * CC);   // x, n-major
  h16*   wq   = (h16*)alloc(sizeof(h16) * (size_t)4 * CC * CC);
  h16*   w1   = (h16*)alloc(sizeof(h16) * (size_t)HIDD * CC);
  h16*   w2   = (h16*)alloc(sizeof(h16) * (size_t)CC * HIDD);
  h16*   ybuf = (h16*)alloc(sizeof(h16) * (size_t)TB * HIDD * NN);   // pre-LIF
  h16*   qs   = (h16*)alloc(sizeof(h16) * (size_t)TB * NH * NPAD * HD);
  h16*   ks   = (h16*)alloc(sizeof(h16) * (size_t)TB * NH * NPAD * HD);
  h16*   vT   = (h16*)alloc(sizeof(h16) * (size_t)TB * NH * HD * NPAD);
  float* ob   = (float*)alloc(sizeof(float) * (size_t)TB * CC * NN); // attn out
  h16*   sbT  = (h16*)alloc(sizeof(h16) * (size_t)TB * NPAD * CC);   // attn spikes
  h16*   x1T  = (h16*)alloc(sizeof(h16) * (size_t)TB * NPAD * CC);   // x1, n-major
  h16*   hT   = (h16*)alloc(sizeof(h16) * (size_t)TB * NPAD * HIDD); // fc1 spikes

  auto cdiv = [](size_t a, size_t b) { return (int)((a + b - 1) / b); };

  // ---- input transpose + weight conversions
  ktransp<<<cdiv((size_t)TB * NPAD * CC, 256), 256, 0, stream>>>(x, xT);
  kcvt<<<cdiv((size_t)4 * CC * CC, 256), 256, 0, stream>>>(qkvp_w, wq, 4 * CC * CC);
  kcvt<<<cdiv((size_t)HIDD * CC, 256), 256, 0, stream>>>(fc1_w, w1, HIDD * CC);
  kcvt<<<cdiv((size_t)CC * HIDD, 256), 256, 0, stream>>>(fc2_w, w2, CC * HIDD);

  const dim3 gC(NPAD / 32, CC / 128, TB);      // (7, 3, 128)
  const int  lifC = cdiv((size_t)BB * CC * NPAD, 256);

  // ---- q, k, v branches: GEMM -> BN+LIF -> spikes (padded head layouts)
  for (int br = 0; br < 3; ++br) {
    kgemm<<<gC, 128, 0, stream>>>(wq + (size_t)br * CC * CC, xT, nullptr, ybuf, CC, CC);
    klif_qkv<<<lifC, 256, 0, stream>>>(ybuf, qkvp_bn + (size_t)br * 4 * CC,
                                       (br == 0) ? qs : ks, vT, out_v,
                                       (br == 2) ? 1 : 0);
  }

  // ---- fused spiking self-attention (exact in f16)
  kattn<<<dim3(13, NH, TB), 32, 0, stream>>>(qs, ks, vT, ob);
  klif_attn<<<lifC, 256, 0, stream>>>(ob, sbT);

  // ---- projection + BN + LIF + attention residual
  kgemm<<<gC, 128, 0, stream>>>(wq + (size_t)3 * CC * CC, sbT, nullptr, ybuf, CC, CC);
  klif_proj<<<lifC, 256, 0, stream>>>(ybuf, qkvp_bn + (size_t)3 * 4 * CC, x, out_x, x1T);

  // ---- MLP: fc1 -> LIF -> fc2 -> LIF (+ residual into out_x)
  kgemm<<<dim3(NPAD / 32, HIDD / 128, TB), 128, 0, stream>>>(w1, x1T, fc1_b, ybuf, HIDD, CC);
  klif_mlp<<<cdiv((size_t)BB * HIDD * NPAD, 256), 256, 0, stream>>>(ybuf, fc1_bn, HIDD, hT, nullptr);
  kgemm<<<gC, 128, 0, stream>>>(w2, hT, fc2_b, ybuf, CC, HIDD);
  klif_mlp<<<lifC, 256, 0, stream>>>(ybuf, fc2_bn, CC, nullptr, out_x);
}
